// ProcessorBlock_8959301780005
// MI455X (gfx1250) — compile-verified, matrix operand order checked by reading, not run on hardware
//
#include <hip/hip_runtime.h>
#include <hip/hip_bf16.h>
#include <math.h>

#define NNODES 40962
#define NEDGES 245760
#define HDIM 256
#define NHEAD 8
#define CDIM 32

typedef unsigned short u16;
typedef __attribute__((ext_vector_type(16))) __bf16 v16bf;
typedef __attribute__((ext_vector_type(4)))  __bf16 v4bf;
typedef __attribute__((ext_vector_type(8)))  float  v8f;
typedef __attribute__((ext_vector_type(4)))  float  v4f;

union Frag {
    v16bf v;
    v4bf  b4[4];
    u16   s[16];
    uint4 q[2];
};

__device__ __forceinline__ u16 f2bf(float f) {
    unsigned u = __float_as_uint(f);
    u += 0x7fffu + ((u >> 16) & 1u);      // round-to-nearest-even
    return (u16)(u >> 16);
}

__device__ __forceinline__ float gelu_tanh(float x) {
    float x3 = x * x * x;
    return 0.5f * x * (1.0f + tanhf(0.7978845608028654f * (x + 0.044715f * x3)));
}

__device__ __forceinline__ void atomicMaxF(float* a, float v) {
    if (v >= 0.f) atomicMax((int*)a, __float_as_int(v));
    else          atomicMin((unsigned int*)a, __float_as_uint(v));
}

// Unconditional 16-element bf16 A fragment (row pointer pre-clamped, always
// safe to read; invalid rows only feed discarded output rows).
__device__ __forceinline__ Frag load_a_frag(const float* __restrict__ arow,
                                            int k0, int k1) {
    v4f f00 = *(const v4f*)(arow + k0);
    v4f f01 = *(const v4f*)(arow + k0 + 4);
    v4f f10 = *(const v4f*)(arow + k1);
    v4f f11 = *(const v4f*)(arow + k1 + 4);
    Frag a;
    a.b4[0] = __builtin_convertvector(f00, v4bf);
    a.b4[1] = __builtin_convertvector(f01, v4bf);
    a.b4[2] = __builtin_convertvector(f10, v4bf);
    a.b4[3] = __builtin_convertvector(f11, v4bf);
    return a;
}

// Guarded variant for K-padded GEMMs (Kreal % 8 == 0; guard uniform per chunk).
__device__ __forceinline__ Frag load_a_frag_guard(const float* __restrict__ arow,
                                                  int k0, int k1, int Kreal) {
    v4f z = {0.f, 0.f, 0.f, 0.f};
    v4f f00 = z, f01 = z, f10 = z, f11 = z;
    if ((k0 + 7) < Kreal) {
        f00 = *(const v4f*)(arow + k0);
        f01 = *(const v4f*)(arow + k0 + 4);
    }
    if ((k1 + 7) < Kreal) {
        f10 = *(const v4f*)(arow + k1);
        f11 = *(const v4f*)(arow + k1 + 4);
    }
    Frag a;
    a.b4[0] = __builtin_convertvector(f00, v4bf);
    a.b4[1] = __builtin_convertvector(f01, v4bf);
    a.b4[2] = __builtin_convertvector(f10, v4bf);
    a.b4[3] = __builtin_convertvector(f11, v4bf);
    return a;
}

// ---------------------------------------------------------------------------
// Pack W[K,256] (f32, row-major) into per-lane bf16 WMMA B fragments.
// Tile (kt,nt): 32 lanes x 16 bf16, lane L holds column nt*16+(L%16),
// elems 0..7  = K kt*32 + (L/16)*8 + j
// elems 8..15 = K kt*32 + 16 + (L/16)*8 + j      (matches ISA 16-bit A layout)
// Zero-filled for k >= Kreal.
// ---------------------------------------------------------------------------
__global__ void pack_b(const float* __restrict__ W, int Kloop, int Kreal,
                       u16* __restrict__ out) {
    int tid = blockIdx.x * blockDim.x + threadIdx.x;
    int total = (Kloop >> 5) * 512;                 // tiles * 32 lanes
    if (tid >= total) return;
    int lane = tid & 31;
    int nt   = (tid >> 5) & 15;
    int kt   = tid >> 9;
    int n     = nt * 16 + (lane & 15);
    int khalf = lane >> 4;
    Frag t;
#pragma unroll
    for (int j = 0; j < 8; ++j) {
        int k0 = kt * 32 + khalf * 8 + j;
        int k1 = kt * 32 + 16 + khalf * 8 + j;
        t.s[j]     = (k0 < Kreal) ? f2bf(W[(size_t)k0 * 256 + n]) : (u16)0;
        t.s[8 + j] = (k1 < Kreal) ? f2bf(W[(size_t)k1 * 256 + n]) : (u16)0;
    }
    uint4* o = (uint4*)(out + (size_t)tid * 16);
    o[0] = t.q[0];
    o[1] = t.q[1];
}

// ---------------------------------------------------------------------------
// Generic GEMM: Out[M,256] = act(A[M,K] @ B + bias + C0)
// block = 256 threads = 8 waves; wave computes 16 rows x 64 cols.
// FULL: Kloop == Kreal (no K padding) -> branch-free A loads.
// ---------------------------------------------------------------------------
template <bool FULL>
__global__ __launch_bounds__(256)
void gemm_wmma(const float* __restrict__ A, int lda, int M, int Kloop, int Kreal,
               const u16* __restrict__ Bp, const float* __restrict__ bias,
               const float* __restrict__ C0, float* __restrict__ Out, int act) {
    int wid  = threadIdx.x >> 5;
    int lane = threadIdx.x & 31;
    int rowBase = (blockIdx.x * 2 + (wid >> 2)) * 16;
    int colT0   = (wid & 3) * 4;                    // first 16-col tile index
    int m     = lane & 15;
    int khalf = lane >> 4;
    int row   = rowBase + m;
    const float* arow = A + (size_t)(row < M ? row : 0) * lda;

    v8f acc[4] = {};
    int nkt = Kloop >> 5;
    for (int kt = 0; kt < nkt; ++kt) {
        int k0 = kt * 32 + khalf * 8;
        Frag a = FULL ? load_a_frag(arow, k0, k0 + 16)
                      : load_a_frag_guard(arow, k0, k0 + 16, Kreal);
#pragma unroll
        for (int t = 0; t < 4; ++t) {
            Frag b;
            const uint4* bp =
                (const uint4*)Bp + (((size_t)kt * 16 + colT0 + t) * 32 + lane) * 2;
            b.q[0] = bp[0];
            b.q[1] = bp[1];
            acc[t] = __builtin_amdgcn_wmma_f32_16x16x32_bf16(
                false, a.v, false, b.v, (short)0, acc[t], false, false);
        }
    }
#pragma unroll
    for (int t = 0; t < 4; ++t) {
        int col = (colT0 + t) * 16 + m;
#pragma unroll
        for (int j = 0; j < 8; ++j) {
            int r = rowBase + j + khalf * 8;
            if (r < M) {
                float v = acc[t][j];
                if (bias) v += bias[col];
                if (C0)   v += C0[(size_t)r * 256 + col];
                if (act == 1) v = gelu_tanh(v);
                Out[(size_t)r * 256 + col] = v;
            }
        }
    }
}

// ---------------------------------------------------------------------------
// Edge-update MLP stage 1:
// u = gelu( e @ Wu1[64:320] + out[src] @ Wu1[0:32] + out[dst] @ Wu1[32:64] + b )
// ---------------------------------------------------------------------------
__global__ __launch_bounds__(256)
void gemm_mlp1(const float* __restrict__ Ef, const float* __restrict__ X,
               const int* __restrict__ src, const int* __restrict__ dst,
               const u16* __restrict__ BpE, const u16* __restrict__ BpS,
               const u16* __restrict__ BpD, const float* __restrict__ bias,
               float* __restrict__ Out, int M) {
    int wid  = threadIdx.x >> 5;
    int lane = threadIdx.x & 31;
    int rowBase = (blockIdx.x * 2 + (wid >> 2)) * 16;
    int colT0   = (wid & 3) * 4;
    int m     = lane & 15;
    int khalf = lane >> 4;
    int row   = rowBase + m;
    bool ok   = row < M;
    const float* arow = Ef + (size_t)(ok ? row : 0) * 256;

    v8f acc[4] = {};
    for (int kt = 0; kt < 8; ++kt) {           // e part, K=256
        int k0 = kt * 32 + khalf * 8;
        Frag a = load_a_frag(arow, k0, k0 + 16);
#pragma unroll
        for (int t = 0; t < 4; ++t) {
            Frag b;
            const uint4* bp =
                (const uint4*)BpE + (((size_t)kt * 16 + colT0 + t) * 32 + lane) * 2;
            b.q[0] = bp[0]; b.q[1] = bp[1];
            acc[t] = __builtin_amdgcn_wmma_f32_16x16x32_bf16(
                false, a.v, false, b.v, (short)0, acc[t], false, false);
        }
    }
    // gathered K=32 parts (src then dst)
    for (int g = 0; g < 2; ++g) {
        const int* idx = g ? dst : src;
        const u16* Bg  = g ? BpD : BpS;
        int node = ok ? idx[row] : 0;
        const float* xr = X + (size_t)node * 32;
        Frag a = load_a_frag(xr, khalf * 8, 16 + khalf * 8);
#pragma unroll
        for (int t = 0; t < 4; ++t) {
            Frag b;
            const uint4* bp =
                (const uint4*)Bg + (((size_t)(colT0 + t)) * 32 + lane) * 2;
            b.q[0] = bp[0]; b.q[1] = bp[1];
            acc[t] = __builtin_amdgcn_wmma_f32_16x16x32_bf16(
                false, a.v, false, b.v, (short)0, acc[t], false, false);
        }
    }
#pragma unroll
    for (int t = 0; t < 4; ++t) {
        int col = (colT0 + t) * 16 + m;
#pragma unroll
        for (int j = 0; j < 8; ++j) {
            int r = rowBase + j + khalf * 8;
            if (r < M)
                Out[(size_t)r * 256 + col] = gelu_tanh(acc[t][j] + bias[col]);
        }
    }
}

// ---------------------------------------------------------------------------
__global__ void fill_f32(float* p, float v, int n) {
    int t = blockIdx.x * blockDim.x + threadIdx.x;
    if (t < n) p[t] = v;
}

// s[e,h] = sum_c lrelu(xl[src]+xr[dst]+ee)*att ; atomic segment max into smax
__global__ void score_kernel(const float* __restrict__ xl, const float* __restrict__ xr,
                             const float* __restrict__ ee, const int* __restrict__ src,
                             const int* __restrict__ dst, const float* __restrict__ att,
                             float* __restrict__ s, float* __restrict__ smax, int E) {
    int t = blockIdx.x * blockDim.x + threadIdx.x;
    if (t >= E * NHEAD) return;
    int e = t >> 3, h = t & 7;
    int si = src[e], di = dst[e];
    const float* pl = xl + (size_t)si * 256 + h * 32;
    const float* pr = xr + (size_t)di * 256 + h * 32;
    const float* pe = ee + (size_t)e * 256 + h * 32;
    const float* pa = att + h * 32;
    float acc = 0.f;
#pragma unroll
    for (int c = 0; c < 32; ++c) {
        float m = pl[c] + pr[c] + pe[c];
        m = (m > 0.f) ? m : 0.2f * m;
        acc += m * pa[c];
    }
    s[t] = acc;
    atomicMaxF(&smax[di * NHEAD + h], acc);
}

__global__ void softmax_kernel(float* __restrict__ s, const float* __restrict__ smax,
                               float* __restrict__ den, const int* __restrict__ dst, int E) {
    int t = blockIdx.x * blockDim.x + threadIdx.x;
    if (t >= E * NHEAD) return;
    int e = t >> 3, h = t & 7;
    float ex = __expf(s[t] - smax[dst[e] * NHEAD + h]);
    s[t] = ex;
    atomicAdd(&den[dst[e] * NHEAD + h], ex);
}

__global__ void scatter_kernel(const float* __restrict__ s, const float* __restrict__ den,
                               const float* __restrict__ xl, const int* __restrict__ src,
                               const int* __restrict__ dst, float* __restrict__ agg, int E) {
    int t = blockIdx.x * blockDim.x + threadIdx.x;
    if (t >= E * NHEAD) return;
    int e = t >> 3, h = t & 7;
    int si = src[e], di = dst[e];
    float a = s[t] / (den[di * NHEAD + h] + 1e-16f);
    const float* pl = xl + (size_t)si * 256 + h * 32;
    float* pa = agg + (size_t)di * 256 + h * 32;
#pragma unroll
    for (int c = 0; c < 32; ++c) atomicAdd(&pa[c], a * pl[c]);
}

// out[n,c] = mean_h agg[n,h,c] + (x @ Wres)[n,c] + bn[c]
__global__ void node_out_kernel(const float* __restrict__ agg, const float* __restrict__ xin,
                                int Kin, const float* __restrict__ Wres,
                                const float* __restrict__ bn, float* __restrict__ out, int N) {
    int t = blockIdx.x * blockDim.x + threadIdx.x;
    if (t >= N * 32) return;
    int n = t >> 5, c = t & 31;
    float s = 0.f;
#pragma unroll
    for (int h = 0; h < NHEAD; ++h) s += agg[(size_t)n * 256 + h * 32 + c];
    s *= 0.125f;
    const float* xp = xin + (size_t)n * Kin;
    for (int k = 0; k < Kin; ++k) s += xp[k] * Wres[k * 32 + c];
    out[t] = s + bn[c];
}

__global__ void layernorm_kernel(const float* __restrict__ in, const float* __restrict__ g,
                                 const float* __restrict__ b, float* __restrict__ out, int N) {
    int n = blockIdx.x * blockDim.x + threadIdx.x;
    if (n >= N) return;
    const float* x = in + (size_t)n * 32;
    float mu = 0.f;
#pragma unroll
    for (int c = 0; c < 32; ++c) mu += x[c];
    mu *= (1.f / 32.f);
    float var = 0.f;
#pragma unroll
    for (int c = 0; c < 32; ++c) { float d = x[c] - mu; var += d * d; }
    var *= (1.f / 32.f);
    float r = rsqrtf(var + 1e-5f);
    float* o = out + (size_t)n * 32;
#pragma unroll
    for (int c = 0; c < 32; ++c) o[c] = (x[c] - mu) * r * g[c] + b[c];
}

// SGFormer kv = sum_n ks ⊗ v, ksum = sum_n ks ; register accumulation,
// one atomicAdd per block per element. thread = (h,d).
__global__ __launch_bounds__(256)
void kv_reduce_kernel(const float* __restrict__ k, const float* __restrict__ v,
                      int N, int chunk, float* __restrict__ kv, float* __restrict__ ksum) {
    __shared__ float kn[256], vn[256], invn[8];
    int t = threadIdx.x, h = t >> 5, d = t & 31;
    float acc[32];
#pragma unroll
    for (int c = 0; c < 32; ++c) acc[c] = 0.f;
    float ksacc = 0.f;
    int start = blockIdx.x * chunk;
    int end = start + chunk;
    if (end > N) end = N;
    for (int n = start; n < end; ++n) {
        kn[t] = k[(size_t)n * 256 + t];
        vn[t] = v[(size_t)n * 256 + t];
        __syncthreads();
        if (d == 0) {
            float ss = 0.f;
#pragma unroll
            for (int c = 0; c < 32; ++c) ss += kn[h * 32 + c] * kn[h * 32 + c];
            invn[h] = 1.f / (sqrtf(ss) + 1e-6f);
        }
        __syncthreads();
        float iv = invn[h];
        float vv = vn[h * 32 + d];
#pragma unroll
        for (int c = 0; c < 32; ++c) acc[c] += kn[h * 32 + c] * iv * vv;
        ksacc += kn[h * 32 + d] * iv;
        __syncthreads();
    }
#pragma unroll
    for (int c = 0; c < 32; ++c) atomicAdd(&kv[((size_t)h * 32 + c) * 32 + d], acc[c]);
    atomicAdd(&ksum[h * 32 + d], ksacc);
}

// final: trans = mean_h (qs@kv + n v)/(qs.ksum + n); y = LN(a x + (1-a) trans)
__global__ __launch_bounds__(128)
void final_kernel(const float* __restrict__ q, const float* __restrict__ v,
                  const float* __restrict__ kv, const float* __restrict__ ksum,
                  const float* __restrict__ x, const float* __restrict__ alpha_p,
                  const float* __restrict__ cg, const float* __restrict__ cb,
                  float* __restrict__ out, int N) {
    int n = blockIdx.x * blockDim.x + threadIdx.x;
    if (n >= N) return;
    float trans[32];
#pragma unroll
    for (int d = 0; d < 32; ++d) trans[d] = 0.f;
    float fn = (float)N;
    for (int h = 0; h < NHEAD; ++h) {
        const float* qp = q + (size_t)n * 256 + h * 32;
        const float* vp = v + (size_t)n * 256 + h * 32;
        float ss = 0.f;
#pragma unroll
        for (int c = 0; c < 32; ++c) ss += qp[c] * qp[c];
        float iv = 1.f / (sqrtf(ss) + 1e-6f);
        float den = fn;
        for (int c = 0; c < 32; ++c) den += qp[c] * iv * ksum[h * 32 + c];
        float rden = 1.f / den;
        for (int d = 0; d < 32; ++d) {
            float num = fn * vp[d];
            for (int c = 0; c < 32; ++c)
                num += qp[c] * iv * kv[((size_t)h * 32 + c) * 32 + d];
            trans[d] += num * rden;
        }
    }
    float alpha = 1.f / (1.f + __expf(-alpha_p[0]));
    float y[32];
    float mu = 0.f;
#pragma unroll
    for (int d = 0; d < 32; ++d) {
        y[d] = alpha * x[(size_t)n * 32 + d] + (1.f - alpha) * trans[d] * 0.125f;
        mu += y[d];
    }
    mu *= (1.f / 32.f);
    float var = 0.f;
#pragma unroll
    for (int d = 0; d < 32; ++d) { float dd = y[d] - mu; var += dd * dd; }
    var *= (1.f / 32.f);
    float r = rsqrtf(var + 1e-5f);
    float* o = out + (size_t)n * 32;
#pragma unroll
    for (int d = 0; d < 32; ++d) o[d] = (y[d] - mu) * r * cg[d] + cb[d];
}

// ===========================================================================
extern "C" void kernel_launch(void* const* d_in, const int* in_sizes, int n_in,
                              void* d_out, int out_size, void* d_ws, size_t ws_size,
                              hipStream_t stream) {
    const float* mesh  = (const float*)d_in[0];
    const int*   eidx  = (const int*)d_in[1];
    const int*   srcI  = eidx;
    const int*   dstI  = eidx + NEDGES;
    const float* eattr = (const float*)d_in[2];
    const float* epW   = (const float*)d_in[3];
    const float* epB   = (const float*)d_in[4];
    const float* Wl0   = (const float*)d_in[5];
    const float* Wr0   = (const float*)d_in[6];
    const float* WlS   = (const float*)d_in[7];
    const float* WrS   = (const float*)d_in[8];
    const float* We0   = (const float*)d_in[9];
    const float* WeS   = (const float*)d_in[10];
    const float* att0  = (const float*)d_in[11];
    const float* attS  = (const float*)d_in[12];
    const float* Wres0 = (const float*)d_in[13];
    const float* WresS = (const float*)d_in[14];
    const float* bn0   = (const float*)d_in[15];
    const float* bnS   = (const float*)d_in[16];
    const float* Wu1_0 = (const float*)d_in[17];
    const float* Wu1S  = (const float*)d_in[18];
    const float* bu1_0 = (const float*)d_in[19];
    const float* bu1S  = (const float*)d_in[20];
    const float* Wu2_0 = (const float*)d_in[21];
    const float* Wu2S  = (const float*)d_in[22];
    const float* bu2_0 = (const float*)d_in[23];
    const float* bu2S  = (const float*)d_in[24];
    const float* lng   = (const float*)d_in[25];
    const float* lnb   = (const float*)d_in[26];
    const float* Wq    = (const float*)d_in[27];
    const float* Wk    = (const float*)d_in[28];
    const float* Wv    = (const float*)d_in[29];
    const float* alphp = (const float*)d_in[30];
    const float* cng   = (const float*)d_in[31];
    const float* cnb   = (const float*)d_in[32];

    size_t off = 0;
    auto alloc = [&](size_t b) -> char* {
        char* p = (char*)d_ws + off;
        off += (b + 255) & ~(size_t)255;
        return p;
    };
    float* eF   = (float*)alloc((size_t)NEDGES * 256 * 4);
    float* buf1 = (float*)alloc((size_t)NEDGES * 256 * 4);   // ee, then u
    float* xlF  = (float*)alloc((size_t)NNODES * 256 * 4);   // later: q
    float* xrF  = (float*)alloc((size_t)NNODES * 256 * 4);   // later: k
    float* aggF = (float*)alloc((size_t)NNODES * 256 * 4);   // later: v
    float* xcur = (float*)alloc((size_t)NNODES * 32 * 4);
    float* xnxt = (float*)alloc((size_t)NNODES * 32 * 4);
    float* sbuf = (float*)alloc((size_t)NEDGES * NHEAD * 4);
    float* smax = (float*)alloc((size_t)NNODES * NHEAD * 4);
    float* denb = (float*)alloc((size_t)NNODES * NHEAD * 4);
    float* kvb  = (float*)alloc(8 * 32 * 32 * 4);
    float* ksb  = (float*)alloc(8 * 32 * 4);
    u16* packArea = (u16*)alloc((size_t)6 << 20);

    size_t poff = 0;
    auto packW = [&](const float* W, int Kloop, int Kreal) -> u16* {
        u16* p = packArea + poff;
        poff += (size_t)Kloop * 256;
        int total = (Kloop / 32) * 512;
        pack_b<<<(total + 255) / 256, 256, 0, stream>>>(W, Kloop, Kreal, p);
        return p;
    };
    auto gemm = [&](const float* A, int lda, int M, int Kloop,
                    const u16* Bp, const float* bias, const float* C0,
                    float* Out, int act) {
        gemm_wmma<true><<<(M + 31) / 32, 256, 0, stream>>>(A, lda, M, Kloop, Kloop,
                                                           Bp, bias, C0, Out, act);
    };

    // ---- pack all weights into WMMA fragment layout -----------------------
    u16 *WlP[4], *WrP[4], *WeP[4], *Wu1eP[3], *Wu1sP[3], *Wu1dP[3], *Wu2P[3];
    u16* epP = packW(epW, 32, 16);
    WlP[0] = packW(Wl0, 256, 256);
    WrP[0] = packW(Wr0, 256, 256);
    WeP[0] = packW(We0, 256, 256);
    for (int i = 1; i < 4; ++i) {
        WlP[i] = packW(WlS + (size_t)(i - 1) * 32 * 256, 32, 32);
        WrP[i] = packW(WrS + (size_t)(i - 1) * 32 * 256, 32, 32);
        WeP[i] = packW(WeS + (size_t)(i - 1) * 256 * 256, 256, 256);
    }
    for (int i = 0; i < 3; ++i) {
        const float* w1 = (i == 0) ? Wu1_0 : Wu1S + (size_t)(i - 1) * 320 * 256;
        Wu1sP[i] = packW(w1, 32, 32);
        Wu1dP[i] = packW(w1 + 32 * 256, 32, 32);
        Wu1eP[i] = packW(w1 + 64 * 256, 256, 256);
        Wu2P[i]  = packW((i == 0) ? Wu2_0 : Wu2S + (size_t)(i - 1) * 256 * 256, 256, 256);
    }
    u16* WqP = packW(Wq, 256, 256);
    u16* WkP = packW(Wk, 256, 256);
    u16* WvP = packW(Wv, 256, 256);

    // ---- edge input projection: e = edge_attr @ ep_W + ep_b (K padded) ----
    gemm_wmma<false><<<(NEDGES + 31) / 32, 256, 0, stream>>>(
        eattr, 16, NEDGES, 32, 16, epP, epB, nullptr, eF, 0);

    // ---- GAT layers -------------------------------------------------------
    const int EH = NEDGES * NHEAD;
    const float* xin = mesh;
    int Kin = 256;
    for (int i = 0; i < 4; ++i) {
        const float* attL  = (i == 0) ? att0  : attS  + (size_t)(i - 1) * 256;
        const float* WresL = (i == 0) ? Wres0 : WresS + (size_t)(i - 1) * 1024;
        const float* bnL   = (i == 0) ? bn0   : bnS   + (size_t)(i - 1) * 32;

        gemm(xin, Kin, NNODES, Kin, WlP[i], nullptr, nullptr, xlF, 0);
        gemm(xin, Kin, NNODES, Kin, WrP[i], nullptr, nullptr, xrF, 0);
        gemm(eF, 256, NEDGES, 256, WeP[i], nullptr, nullptr, buf1, 0);

        hipMemsetAsync(denb, 0, (size_t)NNODES * NHEAD * 4, stream);
        hipMemsetAsync(aggF, 0, (size_t)NNODES * 256 * 4, stream);
        fill_f32<<<(NNODES * NHEAD + 255) / 256, 256, 0, stream>>>(smax, -3.0e38f,
                                                                   NNODES * NHEAD);
        score_kernel<<<(EH + 255) / 256, 256, 0, stream>>>(xlF, xrF, buf1, srcI, dstI,
                                                           attL, sbuf, smax, NEDGES);
        softmax_kernel<<<(EH + 255) / 256, 256, 0, stream>>>(sbuf, smax, denb, dstI,
                                                             NEDGES);
        scatter_kernel<<<(EH + 255) / 256, 256, 0, stream>>>(sbuf, denb, xlF, srcI,
                                                             dstI, aggF, NEDGES);
        node_out_kernel<<<(NNODES * 32 + 255) / 256, 256, 0, stream>>>(
            aggF, xin, Kin, WresL, bnL, xnxt, NNODES);

        if (i < 3) {
            const float* bu1L = (i == 0) ? bu1_0 : bu1S + (size_t)(i - 1) * 256;
            const float* bu2L = (i == 0) ? bu2_0 : bu2S + (size_t)(i - 1) * 256;
            gemm_mlp1<<<(NEDGES + 31) / 32, 256, 0, stream>>>(
                eF, xnxt, srcI, dstI, Wu1eP[i], Wu1sP[i], Wu1dP[i], bu1L, buf1, NEDGES);
            gemm(buf1, 256, NEDGES, 256, Wu2P[i], bu2L, eF, eF, 0);
            layernorm_kernel<<<(NNODES + 255) / 256, 256, 0, stream>>>(
                xnxt, lng + (size_t)i * 32, lnb + (size_t)i * 32, xcur, NNODES);
            xin = xcur;
            Kin = 32;
        }
    }

    // ---- SGFormer global branch (on mesh_enc) -----------------------------
    gemm(mesh, 256, NNODES, 256, WqP, nullptr, nullptr, xlF, 0);   // q
    gemm(mesh, 256, NNODES, 256, WkP, nullptr, nullptr, xrF, 0);   // k
    gemm(mesh, 256, NNODES, 256, WvP, nullptr, nullptr, aggF, 0);  // v
    hipMemsetAsync(kvb, 0, 8 * 32 * 32 * 4, stream);
    hipMemsetAsync(ksb, 0, 8 * 32 * 4, stream);
    int chunk = (NNODES + 127) / 128;
    kv_reduce_kernel<<<128, 256, 0, stream>>>(xrF, aggF, NNODES, chunk, kvb, ksb);
    final_kernel<<<(NNODES + 127) / 128, 128, 0, stream>>>(
        xlF, aggF, kvb, ksb, xnxt, alphp, cng, cnb, (float*)d_out, NNODES);
}
